// Head_80934363726266
// MI455X (gfx1250) — matrix operand order repaired and checked by reading, not compiled, not run on hardware
//
#include <hip/hip_runtime.h>
#include <hip/hip_bf16.h>

// ---------------------------------------------------------------------------
// Causal single-head attention, B=8, T=4096, E=2048, H=128 (fp32 in/out).
// Kernel 1: QKV projection (bf16 WMMA, f32 accum) -> bf16 Q/K in row-major,
//           V stored pre-transposed [B][H][T] so kernel 2 never transposes.
// Kernel 2: flash attention; K/V tiles double-buffered in LDS via
//           global_load_async_to_lds_b128 (ASYNCcnt), bf16 WMMA for QK^T & PV.
// ---------------------------------------------------------------------------

#define BATCH 8
#define TSEQ  4096
#define EDIM  2048
#define HDIM  128

typedef __attribute__((ext_vector_type(16))) __bf16 v16bf;
typedef __attribute__((ext_vector_type(8)))  __bf16 v8bf;
typedef __attribute__((ext_vector_type(8)))  float  v8f;

__device__ __forceinline__ v8bf ld8(const __bf16* p) {
    // caller guarantees 16-byte alignment -> ds_load_b128 / global_load_b128
    return *(const v8bf*)p;
}

__device__ __forceinline__ v16bf cat8(v8bf lo, v8bf hi) {
    v16bf r;
#pragma unroll
    for (int i = 0; i < 8; ++i) { r[i] = lo[i]; r[i + 8] = hi[i]; }
    return r;
}

// A-matrix fragment (16x32 bf16): lane half selects K base 0/16; lane holds
// 16 contiguous K values base..base+15 (two 16B runs, 16B-aligned).
__device__ __forceinline__ v16bf make_afrag(const __bf16* rowp, int base) {
    return cat8(ld8(rowp + base), ld8(rowp + base + 8));
}

// B-matrix fragment (32x16 bf16): lane holds 16 contiguous K values.
__device__ __forceinline__ v16bf make_bfrag(const __bf16* p) {
    return cat8(ld8(p), ld8(p + 8));
}

__device__ __forceinline__ v8f zero8() {
    v8f z;
#pragma unroll
    for (int i = 0; i < 8; ++i) z[i] = 0.0f;
    return z;
}

__device__ __forceinline__ void lds_fence() {
    // per-wave LDS RAW across lanes: keep program order + drain DS counter
    asm volatile("s_wait_dscnt 0" ::: "memory");
}

// Async 16B copy global -> LDS (tracked by ASYNCcnt, no VGPR round trip).
// LDS byte address = low 32 bits of the flat shared-memory pointer.
__device__ __forceinline__ void async_ld_b128(__bf16* lds, const __bf16* gptr) {
    unsigned l = (unsigned)(uintptr_t)lds;
    asm volatile("global_load_async_to_lds_b128 %0, %1, off"
                 :: "v"(l), "v"(gptr)
                 : "memory");
}

#define WAIT_ASYNC(n) asm volatile("s_wait_asynccnt " #n ::: "memory")

// ---------------------------------------------------------------------------
// Kernel 1: out[r, n] = sum_k x[r, k] * W[k, n]   (M=32768, N=128, K=2048)
// grid.x = 256 (M tiles of 128), grid.y = 3 (Q/K/V), block = 256 (8 waves)
// V output (y==2) is stored transposed: Vt[b][h][t].
// ---------------------------------------------------------------------------
__global__ __launch_bounds__(256) void qkv_proj_kernel(
    const float* __restrict__ x,
    const float* __restrict__ Wk,
    const float* __restrict__ Wq,
    const float* __restrict__ Wv,
    __bf16* __restrict__ Qb,
    __bf16* __restrict__ Kb,
    __bf16* __restrict__ Vtb)
{
    const float* W;
    __bf16* Out;
    if (blockIdx.y == 0)      { W = Wq; Out = Qb; }
    else if (blockIdx.y == 1) { W = Wk; Out = Kb; }
    else                      { W = Wv; Out = Vtb; }
    const bool transposed = (blockIdx.y == 2);

    const int rowBase = blockIdx.x * 128;
    const int tid  = threadIdx.x;
    const int wave = tid >> 5;
    const int lane = tid & 31;

    __shared__ __bf16 Xs[128][40];   // x tile, row-major in K   (80B rows, 16B aligned)
    __shared__ __bf16 Wt[128][40];   // W tile transposed [n][k]

    v8f acc[8];
#pragma unroll
    for (int nt = 0; nt < 8; ++nt) acc[nt] = zero8();

    const int arow  = 16 * wave + (lane & 15);
    const int abase = (lane < 16) ? 0 : 16;
    const int bkoff = (lane < 16) ? 0 : 16;

    for (int kc = 0; kc < EDIM; kc += 32) {
        // --- stage x tile (128 x 32 fp32 -> bf16), 8 threads/row, float4 each
        {
            const int r0 = tid >> 3;
            const int c0 = (tid & 7) * 4;
#pragma unroll
            for (int rr = 0; rr < 128; rr += 32) {
                const float4 v = *(const float4*)(x + (long)(rowBase + r0 + rr) * EDIM + kc + c0);
                Xs[r0 + rr][c0 + 0] = (__bf16)v.x;
                Xs[r0 + rr][c0 + 1] = (__bf16)v.y;
                Xs[r0 + rr][c0 + 2] = (__bf16)v.z;
                Xs[r0 + rr][c0 + 3] = (__bf16)v.w;
            }
        }
        // --- stage W tile transposed (32 x 128 fp32 -> Wt[n][k] bf16)
        for (int idx = tid; idx < 32 * 128; idx += 256) {
            const int k = idx >> 7;
            const int n = idx & 127;
            Wt[n][k] = (__bf16)W[(long)(kc + k) * HDIM + n];
        }
        __syncthreads();

        const v16bf a = make_afrag(&Xs[arow][0], abase);
#pragma unroll
        for (int nt = 0; nt < 8; ++nt) {
            const v16bf b = make_bfrag(&Wt[nt * 16 + (lane & 15)][bkoff]);
            acc[nt] = __builtin_amdgcn_wmma_f32_16x16x32_bf16(
                false, a, false, b, (short)0, acc[nt], false, false);
        }
        __syncthreads();
    }

    // store bf16 result
    const int bb = rowBase / TSEQ;   // tiles never straddle a batch (4096 % 128 == 0)
#pragma unroll
    for (int nt = 0; nt < 8; ++nt) {
        const int col = nt * 16 + (lane & 15);
#pragma unroll
        for (int j = 0; j < 8; ++j) {
            const int row = rowBase + 16 * wave + ((lane < 16) ? j : 8 + j);
            if (transposed) {
                const int t = row - bb * TSEQ;
                Out[((long)bb * HDIM + col) * TSEQ + t] = (__bf16)acc[nt][j];
            } else {
                Out[(long)row * HDIM + col] = (__bf16)acc[nt][j];
            }
        }
    }
}

// ---------------------------------------------------------------------------
// Kernel 2: flash attention. grid = (T/64, B), block = 128 (4 waves).
// Each wave owns 16 query rows; block owns a 64-row query tile.
// K/Vt tiles double-buffered in LDS via async-to-LDS copies.
// ---------------------------------------------------------------------------
__global__ __launch_bounds__(128) void flash_attn_kernel(
    const __bf16* __restrict__ Qb,
    const __bf16* __restrict__ Kb,
    const __bf16* __restrict__ Vtb,   // [B][H][T]
    float* __restrict__ out)
{
    const int bidx = blockIdx.y;
    const int qb   = blockIdx.x * 64;
    const int tid  = threadIdx.x;
    const int wave = tid >> 5;
    const int lane = tid & 31;
    const long base  = (long)bidx * TSEQ * HDIM;   // Q/K/out base
    const long baseV = (long)bidx * HDIM * TSEQ;   // Vt base

    __shared__ __bf16 Ks[2][64][136];   // K tile, row-major in head dim (272B rows)
    __shared__ __bf16 Vt[2][128][72];   // V tile [head][key] (144B rows)
    __shared__ __bf16 Ps[4][16][72];    // per-wave P staging (D-layout -> A-layout)

    const int afbase = (lane < 16) ? 0 : 16;
    const int bkoff  = (lane < 16) ? 0 : 16;

    // --- load Q fragments for this wave's 16 query rows (held all kernel)
    v16bf qfrag[4];
    {
        const int qrow = qb + 16 * wave + (lane & 15);
        const __bf16* qp = Qb + base + (long)qrow * HDIM;
#pragma unroll
        for (int c = 0; c < 4; ++c) qfrag[c] = make_afrag(qp + 32 * c, afbase);
    }

    v8f Oacc[8];
#pragma unroll
    for (int nt = 0; nt < 8; ++nt) Oacc[nt] = zero8();
    float mrow[8], lrow[8];
#pragma unroll
    for (int j = 0; j < 8; ++j) { mrow[j] = -INFINITY; lrow[j] = 0.0f; }

    const float scale = 0.08838834764831845f;   // 1/sqrt(128)
    const int nTiles = (qb >> 6) + 1;

    // issue async copies for one 64-key tile into buffer `buf`
    auto issue_tile = [&](int tile, int buf) {
        const int kb = tile * 64;
        // K tile: 64 rows x 16 segs of 16B  -> 8 async ops/thread
#pragma unroll
        for (int s = 0; s < 8; ++s) {
            const int idx = tid + s * 128;
            const int r = idx >> 4, seg = idx & 15;
            async_ld_b128(&Ks[buf][r][seg * 8],
                          Kb + base + (long)(kb + r) * HDIM + seg * 8);
        }
        // Vt tile: 128 head rows x 8 segs of 16B -> 8 async ops/thread
#pragma unroll
        for (int s = 0; s < 8; ++s) {
            const int idx = tid + s * 128;
            const int h = idx >> 3, seg = idx & 7;
            async_ld_b128(&Vt[buf][h][seg * 8],
                          Vtb + baseV + (long)h * TSEQ + kb + seg * 8);
        }
    };

    issue_tile(0, 0);

    for (int it = 0; it < nTiles; ++it) {
        const int buf = it & 1;
        const int kb  = it * 64;
        const bool more = (it + 1) < nTiles;
        if (more) issue_tile(it + 1, buf ^ 1);
        // wait for the *older* batch only (async loads complete in order)
        if (more) { WAIT_ASYNC(16); } else { WAIT_ASYNC(0); }
        __syncthreads();

        // --- S = Q K^T  (4 key sub-tiles x 4 head-dim chunks)
        v8f s[4];
#pragma unroll
        for (int nt = 0; nt < 4; ++nt) s[nt] = zero8();
#pragma unroll
        for (int nt = 0; nt < 4; ++nt) {
            const __bf16* kp = &Ks[buf][nt * 16 + (lane & 15)][bkoff];
#pragma unroll
            for (int c = 0; c < 4; ++c) {
                const v16bf b = make_bfrag(kp + 32 * c);
                s[nt] = __builtin_amdgcn_wmma_f32_16x16x32_bf16(
                    false, qfrag[c], false, b, (short)0, s[nt], false, false);
            }
        }

        // --- scale + causal mask + online softmax update
        float p[4][8];
#pragma unroll
        for (int nt = 0; nt < 4; ++nt) {
#pragma unroll
            for (int j = 0; j < 8; ++j) {
                const int q = qb + 16 * wave + ((lane < 16) ? j : 8 + j);
                const int k = kb + nt * 16 + (lane & 15);
                float v = s[nt][j] * scale;
                if (k > q) v = -INFINITY;
                p[nt][j] = v;
            }
        }
#pragma unroll
        for (int j = 0; j < 8; ++j) {
            float m = p[0][j];
#pragma unroll
            for (int nt = 1; nt < 4; ++nt) m = fmaxf(m, p[nt][j]);
            // reduce across the 16 lanes holding this row (xor masks stay in half)
            m = fmaxf(m, __shfl_xor(m, 1));
            m = fmaxf(m, __shfl_xor(m, 2));
            m = fmaxf(m, __shfl_xor(m, 4));
            m = fmaxf(m, __shfl_xor(m, 8));
            const float mnew = fmaxf(mrow[j], m);
            const float corr = __expf(mrow[j] - mnew);
            mrow[j] = mnew;
            float sum = 0.0f;
#pragma unroll
            for (int nt = 0; nt < 4; ++nt) {
                const float e = __expf(p[nt][j] - mnew);
                p[nt][j] = e;
                sum += e;
            }
            sum += __shfl_xor(sum, 1);
            sum += __shfl_xor(sum, 2);
            sum += __shfl_xor(sum, 4);
            sum += __shfl_xor(sum, 8);
            lrow[j] = lrow[j] * corr + sum;
#pragma unroll
            for (int nt = 0; nt < 8; ++nt) Oacc[nt][j] *= corr;
        }

        // --- transpose P (D-layout -> A-layout) through per-wave LDS
#pragma unroll
        for (int nt = 0; nt < 4; ++nt) {
#pragma unroll
            for (int j = 0; j < 8; ++j) {
                const int r = (lane < 16) ? j : 8 + j;
                Ps[wave][r][nt * 16 + (lane & 15)] = (__bf16)p[nt][j];
            }
        }
        lds_fence();
        v16bf pf[2];
        {
            const __bf16* pp = &Ps[wave][lane & 15][0];
#pragma unroll
            for (int c = 0; c < 2; ++c) pf[c] = make_afrag(pp + 32 * c, afbase);
        }

        // --- O += P V  (8 head sub-tiles x 2 key chunks)
#pragma unroll
        for (int nt = 0; nt < 8; ++nt) {
            const __bf16* vp = &Vt[buf][nt * 16 + (lane & 15)][bkoff];
#pragma unroll
            for (int c = 0; c < 2; ++c) {
                const v16bf b = make_bfrag(vp + 32 * c);
                Oacc[nt] = __builtin_amdgcn_wmma_f32_16x16x32_bf16(
                    false, pf[c], false, b, (short)0, Oacc[nt], false, false);
            }
        }
        __syncthreads();   // protects buf (reused by issue at it+2) and Ps
    }

    // --- normalize and store fp32 output
#pragma unroll
    for (int nt = 0; nt < 8; ++nt) {
        const int col = nt * 16 + (lane & 15);
#pragma unroll
        for (int j = 0; j < 8; ++j) {
            const int row = qb + 16 * wave + ((lane < 16) ? j : 8 + j);
            out[base + (long)row * HDIM + col] = Oacc[nt][j] / lrow[j];
        }
    }
}

// ---------------------------------------------------------------------------

extern "C" void kernel_launch(void* const* d_in, const int* in_sizes, int n_in,
                              void* d_out, int out_size, void* d_ws, size_t ws_size,
                              hipStream_t stream) {
    const float* x  = (const float*)d_in[0];
    const float* Wk = (const float*)d_in[1];
    const float* Wq = (const float*)d_in[2];
    const float* Wv = (const float*)d_in[3];
    float* out = (float*)d_out;

    const long nElem = (long)BATCH * TSEQ * HDIM;   // 4,194,304
    __bf16* Qb  = (__bf16*)d_ws;
    __bf16* Kb  = Qb + nElem;
    __bf16* Vtb = Kb + nElem;                       // [B][H][T]

    // Kernel 1: Q/K/V projections (bf16 out; V transposed)
    dim3 g1((BATCH * TSEQ) / 128, 3);
    qkv_proj_kernel<<<g1, 256, 0, stream>>>(x, Wk, Wq, Wv, Qb, Kb, Vtb);

    // Kernel 2: causal flash attention
    dim3 g2(TSEQ / 64, BATCH);
    flash_attn_kernel<<<g2, 128, 0, stream>>>(Qb, Kb, Vtb, out);
}
